// NTXent_36825049596413
// MI455X (gfx1250) — matrix operand order repaired and checked by reading, not compiled, not run on hardware
//
#include <hip/hip_runtime.h>
#include <hip/hip_bf16.h>

// ---------------------------------------------------------------------------
// NT-Xent loss, MI455X / gfx1250.
// Gram matrix S*S^T (S = [8192,512] row-normalized, bf16) via
// V_WMMA_F32_16X16X32_BF16 with 64x64 wave tiles (1 vmem load per WMMA),
// fused exp/row-sum epilogue, then log-sum reduce.
// ---------------------------------------------------------------------------

typedef __attribute__((ext_vector_type(8)))  __bf16 v8bf;
typedef __attribute__((ext_vector_type(16))) __bf16 v16bf;
typedef __attribute__((ext_vector_type(8)))  float  v8f;

#define NTX_N     4096
#define NTX_ROWS  8192          // 2N
#define NTX_DIM   512
#define TEMP_INV  2.0f          // 1 / 0.5
#define EXP_SELF  7.38905609893065f   // exp(1/TEMP)

__device__ __forceinline__ unsigned short ntx_f2bf(float f) {
    unsigned int u = __float_as_uint(f);
    u += 0x7FFFu + ((u >> 16) & 1u);      // round-to-nearest-even
    return (unsigned short)(u >> 16);
}
__device__ __forceinline__ float ntx_bf2f(unsigned short h) {
    return __uint_as_float((unsigned int)h << 16);
}

// --------------------------------------------------------------------------
// Kernel 1: row-normalize x|y (f32) -> S (bf16).  One wave per row.
// --------------------------------------------------------------------------
__global__ void __launch_bounds__(256)
ntx_normalize_kernel(const float* __restrict__ x, const float* __restrict__ y,
                     unsigned short* __restrict__ Sb) {
    const int wave = threadIdx.x >> 5;
    const int lane = threadIdx.x & 31;
    const int row  = blockIdx.x * 8 + wave;
    const float* src = (row < NTX_N) ? (x + (size_t)row * NTX_DIM)
                                     : (y + (size_t)(row - NTX_N) * NTX_DIM);
    float v[16];
    float ss = 0.f;
#pragma unroll
    for (int t = 0; t < 16; ++t) {
        v[t] = src[lane + 32 * t];
        ss += v[t] * v[t];
    }
#pragma unroll
    for (int m = 1; m < 32; m <<= 1) ss += __shfl_xor(ss, m, 32);
    const float rn = rsqrtf(ss);
    unsigned short* dst = Sb + (size_t)row * NTX_DIM;
#pragma unroll
    for (int t = 0; t < 16; ++t) dst[lane + 32 * t] = ntx_f2bf(v[t] * rn);
}

// --------------------------------------------------------------------------
// Kernel 2: Gram tiles via WMMA + fused exp/row-sum epilogue.
// Block = 256 thr = 8 waves (2x4). Wave tile = 64x64 -> block tile 128x256.
// Per 32-K step: 4 A frags + 4 B frags (16 b128 loads) feed 16 WMMAs.
// --------------------------------------------------------------------------
__global__ void __launch_bounds__(256)
ntx_gram_expsum_kernel(const unsigned short* __restrict__ Sb,
                       float* __restrict__ rowsum) {
    const int lane = threadIdx.x & 31;
    const int wave = threadIdx.x >> 5;
    const int wm = wave >> 2;                   // 0..1
    const int wn = wave & 3;                    // 0..3
    const int R0 = blockIdx.y * 128 + wm * 64;
    const int C0 = blockIdx.x * 256 + wn * 64;
    const int l15  = lane & 15;
    const int half = lane >> 4;

    // A fragment source: row R0+(lane&15)+16r; contiguous K kk+16h .. kk+16h+15
    const unsigned short* pa = Sb + (size_t)(R0 + l15) * NTX_DIM + half * 16;
    // B fragment source: row C0+(lane&15)+16c; contiguous K kk+16h .. kk+16h+15
    const unsigned short* pb = Sb + (size_t)(C0 + l15) * NTX_DIM + half * 16;

    v8f acc[4][4];
#pragma unroll
    for (int r = 0; r < 4; ++r)
#pragma unroll
        for (int c = 0; c < 4; ++c) acc[r][c] = (v8f){};

#pragma unroll 1
    for (int kk = 0; kk < NTX_DIM; kk += 32) {
        v16bf a[4], b[4];
#pragma unroll
        for (int r = 0; r < 4; ++r) {
            const unsigned short* p = pa + kk + (size_t)r * 16 * NTX_DIM;
            v8bf lo = *(const v8bf*)(p);
            v8bf hi = *(const v8bf*)(p + 8);
            a[r] = __builtin_shufflevector(lo, hi,
                     0,1,2,3,4,5,6,7,8,9,10,11,12,13,14,15);
        }
#pragma unroll
        for (int c = 0; c < 4; ++c)
            b[c] = *(const v16bf*)(pb + kk + (size_t)c * 16 * NTX_DIM);

#pragma unroll
        for (int r = 0; r < 4; ++r)
#pragma unroll
            for (int c = 0; c < 4; ++c)
                acc[r][c] = __builtin_amdgcn_wmma_f32_16x16x32_bf16(
                    false, a[r], false, b[c], (short)0, acc[r][c],
                    false, false);
    }

    // Epilogue: e = sum_cols exp(2*dot); reduce across the 16-lane group;
    // C layout: VGPR j of a 16x16 acc holds row j + 8*(lane/16), col lane%16.
#pragma unroll
    for (int r = 0; r < 4; ++r) {
        const int Ra = R0 + r * 16;
#pragma unroll
        for (int j = 0; j < 8; ++j) {
            float e = __expf(TEMP_INV * acc[r][0][j])
                    + __expf(TEMP_INV * acc[r][1][j])
                    + __expf(TEMP_INV * acc[r][2][j])
                    + __expf(TEMP_INV * acc[r][3][j]);
            e += __shfl_xor(e, 1, 32);
            e += __shfl_xor(e, 2, 32);
            e += __shfl_xor(e, 4, 32);
            e += __shfl_xor(e, 8, 32);
            if ((lane & 15) == 0)               // lanes 0 and 16
                atomicAdd(&rowsum[Ra + j + (half << 3)], e);
        }
    }
}

// --------------------------------------------------------------------------
// Kernel 3: positive-pair dots d_i = <S[i], S[i^4096]> (bf16 -> f32 acc).
// One wave per row.
// --------------------------------------------------------------------------
__global__ void __launch_bounds__(256)
ntx_posdot_kernel(const unsigned short* __restrict__ Sb,
                  float* __restrict__ posdot) {
    const int wave = threadIdx.x >> 5;
    const int lane = threadIdx.x & 31;
    const int row  = blockIdx.x * 8 + wave;
    const unsigned short* pi = Sb + (size_t)row * NTX_DIM;
    const unsigned short* pj = Sb + (size_t)(row ^ NTX_N) * NTX_DIM;
    float acc = 0.f;
#pragma unroll
    for (int t = 0; t < 16; ++t) {
        const int idx = lane + 32 * t;
        acc = fmaf(ntx_bf2f(pi[idx]), ntx_bf2f(pj[idx]), acc);
    }
#pragma unroll
    for (int m = 1; m < 32; m <<= 1) acc += __shfl_xor(acc, m, 32);
    if (lane == 0) posdot[row] = acc;
}

// --------------------------------------------------------------------------
// Kernel 4: loss = ( sum_i [ log(rowsum_i - e^2) - 2*posdot_i ] ) / 8192
// --------------------------------------------------------------------------
__global__ void __launch_bounds__(256)
ntx_final_reduce_kernel(const float* __restrict__ rowsum,
                        const float* __restrict__ posdot,
                        float* __restrict__ out) {
    __shared__ float sm[256];
    float s = 0.f;
    for (int i = threadIdx.x; i < NTX_ROWS; i += 256)
        s += logf(rowsum[i] - EXP_SELF) - TEMP_INV * posdot[i];
    sm[threadIdx.x] = s;
    __syncthreads();
    for (int w = 128; w > 0; w >>= 1) {
        if (threadIdx.x < w) sm[threadIdx.x] += sm[threadIdx.x + w];
        __syncthreads();
    }
    if (threadIdx.x == 0) out[0] = sm[0] / (float)NTX_ROWS;
}

// --------------------------------------------------------------------------
extern "C" void kernel_launch(void* const* d_in, const int* in_sizes, int n_in,
                              void* d_out, int out_size, void* d_ws, size_t ws_size,
                              hipStream_t stream) {
    (void)in_sizes; (void)n_in; (void)out_size; (void)ws_size;
    const float* x = (const float*)d_in[0];
    const float* y = (const float*)d_in[1];
    float* out = (float*)d_out;

    // Workspace layout: [ S bf16 : 8 MB ][ rowsum : 32 KB ][ posdot : 32 KB ]
    unsigned short* Sb = (unsigned short*)d_ws;
    float* rowsum = (float*)((char*)d_ws + (size_t)NTX_ROWS * NTX_DIM * sizeof(unsigned short));
    float* posdot = rowsum + NTX_ROWS;

    // rowsum is accumulated with atomics -> must be zeroed every launch.
    hipMemsetAsync(rowsum, 0, NTX_ROWS * sizeof(float), stream);

    ntx_normalize_kernel<<<NTX_ROWS / 8, 256, 0, stream>>>(x, y, Sb);

    dim3 grid(NTX_ROWS / 256, NTX_ROWS / 128);  // (col blocks, row blocks)
    ntx_gram_expsum_kernel<<<grid, 256, 0, stream>>>(Sb, rowsum);

    ntx_posdot_kernel<<<NTX_ROWS / 8, 256, 0, stream>>>(Sb, posdot);

    ntx_final_reduce_kernel<<<1, 256, 0, stream>>>(rowsum, posdot, out);
}
